// PQAdapter_35373350649918
// MI455X (gfx1250) — matrix operand order repaired and checked by reading, not compiled, not run on hardware
//
#include <hip/hip_runtime.h>
#include <hip/hip_bf16.h>
#include <math.h>

// ---------------------------------------------------------------------------
// Problem constants (from reference)
// ---------------------------------------------------------------------------
#define B_    32
#define M_    4096
#define DIM_  768
#define HID_  128
#define H2_   64
#define GRID_ 15
#define P_    225      // GRID*GRID
#define IMG_  240
#define TOPK_ 10
#define EPS_  1e-5f
#define K1_   (DIM_ * 9)   // conv1 implicit-GEMM K = 6912
#define SQS_  776          // LDS row stride (bf16) for staged B tile, 16B aligned, bank-skewed

typedef __bf16 bf16_t;
typedef __attribute__((ext_vector_type(16))) __bf16 v16bf;
typedef __attribute__((ext_vector_type(8)))  __bf16 v8bf;
typedef __attribute__((ext_vector_type(8)))  float  v8f;

// ---------------------------------------------------------------------------
// 1. L2-normalize tokens and convert to bf16 (one block per token, 768 elems)
// ---------------------------------------------------------------------------
__global__ void k_normalize(const float* __restrict__ q, const float* __restrict__ p,
                            bf16_t* __restrict__ qn, bf16_t* __restrict__ pn) {
  __shared__ float red[256];
  int id = blockIdx.x;
  const float* src;
  bf16_t* dst;
  if (id < B_ * P_) { src = q + (size_t)id * DIM_; dst = qn + (size_t)id * DIM_; }
  else { int j = id - B_ * P_; src = p + (size_t)j * DIM_; dst = pn + (size_t)j * DIM_; }
  int t = threadIdx.x;
  float v0 = src[t], v1 = src[t + 256], v2 = src[t + 512];
  red[t] = v0 * v0 + v1 * v1 + v2 * v2;
  __syncthreads();
  for (int off = 128; off > 0; off >>= 1) {
    if (t < off) red[t] += red[t + off];
    __syncthreads();
  }
  float inv = 1.0f / fmaxf(sqrtf(red[0]), 1e-12f);
  dst[t]       = (bf16_t)(v0 * inv);
  dst[t + 256] = (bf16_t)(v1 * inv);
  dst[t + 512] = (bf16_t)(v2 * inv);
}

// ---------------------------------------------------------------------------
// 2. generic f32 -> bf16 convert (for conv1 weights)
// ---------------------------------------------------------------------------
__global__ void k_f32_to_bf16(const float* __restrict__ s, bf16_t* __restrict__ d, int n) {
  int i = blockIdx.x * blockDim.x + threadIdx.x;
  if (i < n) d[i] = (bf16_t)s[i];
}

// ---------------------------------------------------------------------------
// 3. Fused similarity GEMM + row argmax, WMMA bf16 -> f32.
//    B tile (qn^T, 16 cols x 768 K) staged once in LDS and shared by all 8
//    waves; each wave sweeps two 16-row M-tiles per iteration (A fragments
//    streamed from global), sharing one LDS B-fragment read per K-chunk.
//    C/D layout (ISA 7.12.2): lane holds column p = lane&15; c[r] is row
//    m = r + 8*(lane>>4). Halves merged via shfl_xor(16), waves via LDS.
// ---------------------------------------------------------------------------
__global__ void k_sim_argmax(const bf16_t* __restrict__ qn, const bf16_t* __restrict__ pn,
                             float* __restrict__ maxsim, int* __restrict__ argmax) {
  __shared__ __align__(16) bf16_t sq[16 * SQS_];
  __shared__ float sv[8][16];
  __shared__ int   si[8][16];
  const int b    = blockIdx.y;
  const int p0   = blockIdx.x * 16;
  const int tid  = threadIdx.x;
  const int wave = tid >> 5;
  const int lane = tid & 31;
  const int hi16 = lane >> 4;
  const int lm   = lane & 15;

  // stage B tile (16 columns x 768 K, bf16) into LDS with skewed stride
  for (int idx = tid; idx < 16 * (DIM_ / 8); idx += 256) {
    int col = idx / (DIM_ / 8);
    int ko  = (idx % (DIM_ / 8)) * 8;
    int pc  = p0 + col; if (pc > P_ - 1) pc = P_ - 1;
    *(v8bf*)(sq + col * SQS_ + ko) =
        *(const v8bf*)(qn + (size_t)(b * P_ + pc) * DIM_ + ko);
  }
  __syncthreads();

  const int kbA = hi16 ? 16 : 0;     // A-fragment K-base within 32-chunk
  const int kbB = hi16 ? 16 : 0;     // B-fragment K-base within 32-chunk
  const bf16_t* brow = sq + lm * SQS_ + kbB;

  float best = -1e30f;
  int   bidx = 0;

  // each wave: two 16-row M-tiles per iteration, stride 256 rows
  for (int m0 = wave * 32; m0 < M_; m0 += 256) {
    const bf16_t* arow0 = pn + (size_t)(b * M_ + m0 + lm) * DIM_;
    const bf16_t* arow1 = arow0 + (size_t)16 * DIM_;
    if (m0 + 256 < M_)
      __builtin_prefetch(pn + (size_t)(b * M_ + m0 + 256 + lm) * DIM_, 0, 0);
    v8f c0 = {};
    v8f c1 = {};
#pragma unroll 4
    for (int k0 = 0; k0 < DIM_; k0 += 32) {
      v16bf bfrag = *(const v16bf*)(brow + k0);
      v8bf a0lo = *(const v8bf*)(arow0 + k0 + kbA);
      v8bf a0hi = *(const v8bf*)(arow0 + k0 + kbA + 8);
      v16bf a0 = __builtin_shufflevector(a0lo, a0hi,
                   0, 1, 2, 3, 4, 5, 6, 7, 8, 9, 10, 11, 12, 13, 14, 15);
      c0 = __builtin_amdgcn_wmma_f32_16x16x32_bf16(false, a0, false, bfrag,
                                                   (short)0, c0, false, false);
      v8bf a1lo = *(const v8bf*)(arow1 + k0 + kbA);
      v8bf a1hi = *(const v8bf*)(arow1 + k0 + kbA + 8);
      v16bf a1 = __builtin_shufflevector(a1lo, a1hi,
                   0, 1, 2, 3, 4, 5, 6, 7, 8, 9, 10, 11, 12, 13, 14, 15);
      c1 = __builtin_amdgcn_wmma_f32_16x16x32_bf16(false, a1, false, bfrag,
                                                   (short)0, c1, false, false);
    }
#pragma unroll
    for (int r = 0; r < 8; ++r) {
      float v = c0[r];
      int   m = m0 + r + hi16 * 8;
      if (v > best) { best = v; bidx = m; }   // m ascends -> keeps first max
    }
#pragma unroll
    for (int r = 0; r < 8; ++r) {
      float v = c1[r];
      int   m = m0 + 16 + r + hi16 * 8;
      if (v > best) { best = v; bidx = m; }
    }
  }
  // merge the two lane halves (same column, disjoint row sets)
  float ov = __shfl_xor(best, 16, 32);
  int   oi = __shfl_xor(bidx, 16, 32);
  if (ov > best || (ov == best && oi < bidx)) { best = ov; bidx = oi; }
  if (hi16 == 0) { sv[wave][lm] = best; si[wave][lm] = bidx; }
  __syncthreads();
  if (tid < 16) {
    float bv = sv[0][tid]; int bi = si[0][tid];
    for (int w = 1; w < 8; ++w) {
      float v = sv[w][tid]; int i2 = si[w][tid];
      if (v > bv || (v == bv && i2 < bi)) { bv = v; bi = i2; }
    }
    int pp = p0 + tid;
    if (pp < P_) { maxsim[b * P_ + pp] = bv; argmax[b * P_ + pp] = bi; }
  }
}

// ---------------------------------------------------------------------------
// 4. context / aligned / indices / residual + sbn-normalized transposed bf16
//    feature map for the seg head.  One block per (p, b) token.
// ---------------------------------------------------------------------------
__global__ void k_context(const float* __restrict__ q, const float* __restrict__ pr,
                          const int* __restrict__ layer_idx,
                          const float* __restrict__ sg, const float* __restrict__ sb,
                          const float* __restrict__ sm, const float* __restrict__ svv,
                          const float* __restrict__ maxsim, const int* __restrict__ argmax,
                          float* __restrict__ ctx_out, float* __restrict__ aligned_out,
                          float* __restrict__ idx_out, float* __restrict__ resid_out,
                          bf16_t* __restrict__ cm) {
  int p = blockIdx.x, b = blockIdx.y, t = threadIdx.x;
  int tok = b * P_ + p;
  int idx = argmax[tok];
  int li = layer_idx[0]; li = li < 0 ? 0 : (li > 2 ? 2 : li);
  const float* qrow = q + (size_t)tok * DIM_;
  const float* arow = pr + (size_t)(b * M_ + idx) * DIM_;
  for (int d = t; d < DIM_; d += 256) {
    float qv = qrow[d], av = arow[d];
    float cv = qv + fabsf(qv - av);                  // BETA = 1
    ctx_out[(size_t)tok * DIM_ + d]     = cv;
    aligned_out[(size_t)tok * DIM_ + d] = av;
    float s  = sg[li * DIM_ + d] * rsqrtf(svv[li * DIM_ + d] + EPS_);
    float bn = (cv - sm[li * DIM_ + d]) * s + sb[li * DIM_ + d];
    cm[((size_t)b * DIM_ + d) * P_ + p] = (bf16_t)bn;  // [B][DIM][15*15]
  }
  if (t == 0) {
    idx_out[tok]   = (float)idx;
    resid_out[tok] = 0.5f * (1.0f - maxsim[tok]);
  }
}

// ---------------------------------------------------------------------------
// 5a. explicit im2col (bf16) for conv1: im[b][p][k], k contiguous
// ---------------------------------------------------------------------------
__global__ void k_im2col(const bf16_t* __restrict__ cm, bf16_t* __restrict__ im) {
  size_t i = (size_t)blockIdx.x * 256 + threadIdx.x;
  const size_t total = (size_t)B_ * P_ * K1_;
  if (i >= total) return;
  int k = (int)(i % K1_);
  size_t tmp = i / K1_;
  int p = (int)(tmp % P_);
  int b = (int)(tmp / P_);
  int ci = k / 9, rem = k - ci * 9;
  int ky = rem / 3, kx = rem - ky * 3;
  int y = p / GRID_, x = p % GRID_;
  int yy = y + ky - 1, xx = x + kx - 1;
  bf16_t v = (bf16_t)0.0f;
  if (yy >= 0 && yy < GRID_ && xx >= 0 && xx < GRID_)
    v = cm[((size_t)b * DIM_ + ci) * P_ + yy * GRID_ + xx];
  im[i] = v;
}

// ---------------------------------------------------------------------------
// 5b. conv1 (3x3, 768 -> 128) as GEMM over im2col, WMMA, fused BN+ReLU.
//     A = weights [128 x 6912] bf16, B = im2col rows; one wave per block.
// ---------------------------------------------------------------------------
__global__ void k_conv1_wmma(const bf16_t* __restrict__ w1b, const bf16_t* __restrict__ im,
                             const float* __restrict__ g, const float* __restrict__ bb,
                             const float* __restrict__ mm, const float* __restrict__ vv,
                             float* __restrict__ h1) {
  const int p0 = blockIdx.x * 16;
  const int o0 = blockIdx.y * 16;
  const int b  = blockIdx.z;
  const int lane = threadIdx.x;
  const int hi16 = lane >> 4;
  const int lm   = lane & 15;
  int pc = p0 + lm; if (pc > P_ - 1) pc = P_ - 1;
  const int kbA = hi16 ? 16 : 0;
  const int kbB = hi16 ? 16 : 0;
  const bf16_t* wrow = w1b + (size_t)(o0 + lm) * K1_;
  const bf16_t* brow = im + (size_t)(b * P_ + pc) * K1_ + kbB;
  v8f c = {};
#pragma unroll 4
  for (int k0 = 0; k0 < K1_; k0 += 32) {
    v8bf alo = *(const v8bf*)(wrow + k0 + kbA);
    v8bf ahi = *(const v8bf*)(wrow + k0 + kbA + 8);
    v16bf a = __builtin_shufflevector(alo, ahi,
                0, 1, 2, 3, 4, 5, 6, 7, 8, 9, 10, 11, 12, 13, 14, 15);
    v16bf bfrag = *(const v16bf*)(brow + k0);
    c = __builtin_amdgcn_wmma_f32_16x16x32_bf16(false, a, false, bfrag,
                                                (short)0, c, false, false);
  }
  int pp = p0 + lm;
  if (pp < P_) {
#pragma unroll
    for (int r = 0; r < 8; ++r) {
      int o = o0 + r + hi16 * 8;
      float s  = g[o] * rsqrtf(vv[o] + EPS_);
      float hv = (c[r] - mm[o]) * s + bb[o];
      h1[((size_t)(b * HID_ + o)) * P_ + pp] = fmaxf(hv, 0.0f);
    }
  }
}

// ---------------------------------------------------------------------------
// 6. convT1 2x2 s2 (128->128), 15 -> 30
// ---------------------------------------------------------------------------
__global__ void k_convT1(const float* __restrict__ h1, const float* __restrict__ wt1,
                         const float* __restrict__ bt1, float* __restrict__ out) {
  int i = blockIdx.x * blockDim.x + threadIdx.x;
  const int total = B_ * HID_ * 30 * 30;
  if (i >= total) return;
  int ox = i % 30; int tmp = i / 30;
  int oy = tmp % 30; tmp /= 30;
  int o = tmp % HID_; int b = tmp / HID_;
  int y = oy >> 1, ii = oy & 1, x = ox >> 1, jj = ox & 1;
  float acc = bt1[o];
  const float* hrow = h1 + (size_t)b * HID_ * P_ + y * GRID_ + x;
  const float* w = wt1 + (size_t)o * 4 + ii * 2 + jj;   // [c][o][i][j]
  for (int ci = 0; ci < HID_; ++ci)
    acc += hrow[(size_t)ci * P_] * w[(size_t)ci * HID_ * 4];
  out[i] = acc;
}

// ---------------------------------------------------------------------------
// 7. conv2 3x3 (128->64) on 30x30, fused BN+ReLU
// ---------------------------------------------------------------------------
__global__ void k_conv2(const float* __restrict__ in, const float* __restrict__ w2,
                        const float* __restrict__ g, const float* __restrict__ bb,
                        const float* __restrict__ mm, const float* __restrict__ vv,
                        float* __restrict__ out) {
  int i = blockIdx.x * blockDim.x + threadIdx.x;
  const int total = B_ * H2_ * 900;
  if (i >= total) return;
  int p = i % 900; int tmp = i / 900;
  int o = tmp % H2_; int b = tmp / H2_;
  int y = p / 30, x = p % 30;
  float acc = 0.f;
  const float* inb  = in + (size_t)b * HID_ * 900;
  const float* wrow = w2 + (size_t)o * HID_ * 9;
  for (int ci = 0; ci < HID_; ++ci) {
    const float* ip = inb + (size_t)ci * 900;
    const float* wp = wrow + ci * 9;
#pragma unroll
    for (int ky = 0; ky < 3; ++ky) {
      int yy = y + ky - 1;
      if (yy < 0 || yy >= 30) continue;
#pragma unroll
      for (int kx = 0; kx < 3; ++kx) {
        int xx = x + kx - 1;
        if (xx < 0 || xx >= 30) continue;
        acc += ip[yy * 30 + xx] * wp[ky * 3 + kx];
      }
    }
  }
  float s = g[o] * rsqrtf(vv[o] + EPS_);
  out[i] = fmaxf((acc - mm[o]) * s + bb[o], 0.f);
}

// ---------------------------------------------------------------------------
// 8. convT2 2x2 s2 (64->64), 30 -> 60
// ---------------------------------------------------------------------------
__global__ void k_convT2(const float* __restrict__ h2, const float* __restrict__ wt2,
                         const float* __restrict__ bt2, float* __restrict__ out) {
  int i = blockIdx.x * blockDim.x + threadIdx.x;
  const int total = B_ * H2_ * 60 * 60;
  if (i >= total) return;
  int ox = i % 60; int tmp = i / 60;
  int oy = tmp % 60; tmp /= 60;
  int o = tmp % H2_; int b = tmp / H2_;
  int y = oy >> 1, ii = oy & 1, x = ox >> 1, jj = ox & 1;
  float acc = bt2[o];
  const float* hrow = h2 + (size_t)b * H2_ * 900 + y * 30 + x;
  const float* w = wt2 + (size_t)o * 4 + ii * 2 + jj;   // [c][o][i][j]
  for (int ci = 0; ci < H2_; ++ci)
    acc += hrow[(size_t)ci * 900] * w[(size_t)ci * H2_ * 4];
  out[i] = acc;
}

// ---------------------------------------------------------------------------
// 9. conv3 1x1 (64->2) + bias, also emits patch_map = softmax(low)[:,1]
// ---------------------------------------------------------------------------
__global__ void k_conv3(const float* __restrict__ in, const float* __restrict__ w3,
                        const float* __restrict__ b3, float* __restrict__ low,
                        float* __restrict__ patch) {
  int i = blockIdx.x * blockDim.x + threadIdx.x;
  if (i >= B_ * 3600) return;
  int p = i % 3600, b = i / 3600;
  const float* inb = in + (size_t)b * H2_ * 3600 + p;
  float a0 = b3[0], a1 = b3[1];
  for (int ci = 0; ci < H2_; ++ci) {
    float v = inb[(size_t)ci * 3600];
    a0 += v * w3[ci];
    a1 += v * w3[H2_ + ci];
  }
  low[(size_t)b * 2 * 3600 + p]        = a0;
  low[(size_t)b * 2 * 3600 + 3600 + p] = a1;
  patch[(size_t)b * 3600 + p] = 1.0f / (1.0f + expf(-(a1 - a0)));  // softmax ch1
}

// ---------------------------------------------------------------------------
// 10. bilinear resize 60 -> 240 (half-pixel, align_corners=False)
// ---------------------------------------------------------------------------
__global__ void k_resize(const float* __restrict__ low, float* __restrict__ seg) {
  int i = blockIdx.x * blockDim.x + threadIdx.x;
  const int total = B_ * 2 * IMG_ * IMG_;
  if (i >= total) return;
  int x = i % IMG_; int tmp = i / IMG_;
  int y = tmp % IMG_; tmp /= IMG_;
  int cc = tmp & 1; int b = tmp >> 1;
  float sy = (y + 0.5f) * 0.25f - 0.5f;
  float sx = (x + 0.5f) * 0.25f - 0.5f;
  float fy = floorf(sy), fx = floorf(sx);
  float wy = sy - fy, wx = sx - fx;
  int y0 = (int)fy, x0 = (int)fx;
  int y1 = y0 + 1, x1 = x0 + 1;
  y0 = min(max(y0, 0), 59); y1 = min(max(y1, 0), 59);
  x0 = min(max(x0, 0), 59); x1 = min(max(x1, 0), 59);
  const float* L = low + ((size_t)b * 2 + cc) * 3600;
  float v00 = L[y0 * 60 + x0], v01 = L[y0 * 60 + x1];
  float v10 = L[y1 * 60 + x0], v11 = L[y1 * 60 + x1];
  seg[i] = v00 * (1 - wy) * (1 - wx) + v01 * (1 - wy) * wx
         + v10 * wy * (1 - wx)       + v11 * wy * wx;
}

// ---------------------------------------------------------------------------
// 11. avg-pool + top-k(10) residual pool -> pooled features. One block/batch.
// ---------------------------------------------------------------------------
__global__ void k_pool_topk(const float* __restrict__ ctx, const float* __restrict__ resid,
                            float* __restrict__ pooled) {
  __shared__ float sval[256];
  __shared__ int   sidx[256];
  __shared__ int   topi[TOPK_];
  int b = blockIdx.x, t = threadIdx.x;
  float rv = (t < P_) ? resid[b * P_ + t] : -1e30f;
  for (int k = 0; k < TOPK_; ++k) {
    sval[t] = rv; sidx[t] = t;
    __syncthreads();
    for (int off = 128; off > 0; off >>= 1) {
      if (t < off) {
        if (sval[t + off] > sval[t] ||
            (sval[t + off] == sval[t] && sidx[t + off] < sidx[t])) {
          sval[t] = sval[t + off]; sidx[t] = sidx[t + off];
        }
      }
      __syncthreads();
    }
    if (t == 0) topi[k] = sidx[0];
    __syncthreads();
    if (t == sidx[0]) rv = -1e30f;
    __syncthreads();
  }
  for (int d = t; d < DIM_; d += 256) {
    float s = 0.f;
    for (int p = 0; p < P_; ++p) s += ctx[(size_t)(b * P_ + p) * DIM_ + d];
    float avg = s / (float)P_;
    float ts = 0.f;
    for (int k = 0; k < TOPK_; ++k) ts += ctx[(size_t)(b * P_ + topi[k]) * DIM_ + d];
    pooled[b * DIM_ + d] = 0.5f * avg + 0.5f * (ts / (float)TOPK_);
  }
}

// ---------------------------------------------------------------------------
// 12. global MLP head (768 -> 128 -> 64 -> 2). One block/batch.
// ---------------------------------------------------------------------------
__global__ void k_mlp(const float* __restrict__ pooled,
                      const float* __restrict__ g1, const float* __restrict__ g1g,
                      const float* __restrict__ g1b, const float* __restrict__ g1m,
                      const float* __restrict__ g1v,
                      const float* __restrict__ g2, const float* __restrict__ g2g,
                      const float* __restrict__ g2b, const float* __restrict__ g2m,
                      const float* __restrict__ g2v,
                      const float* __restrict__ g3,
                      float* __restrict__ gl, float* __restrict__ logit,
                      float* __restrict__ prob) {
  __shared__ float s1[HID_];
  __shared__ float s2[H2_];
  __shared__ float sgl[2];
  int b = blockIdx.x, t = threadIdx.x;
  const float* x = pooled + b * DIM_;
  if (t < HID_) {
    float acc = 0.f;
    const float* w = g1 + (size_t)t * DIM_;
    for (int d = 0; d < DIM_; ++d) acc += x[d] * w[d];
    float s = g1g[t] * rsqrtf(g1v[t] + EPS_);
    s1[t] = fmaxf((acc - g1m[t]) * s + g1b[t], 0.f);
  }
  __syncthreads();
  if (t < H2_) {
    float acc = 0.f;
    const float* w = g2 + (size_t)t * HID_;
    for (int d = 0; d < HID_; ++d) acc += s1[d] * w[d];
    float s = g2g[t] * rsqrtf(g2v[t] + EPS_);
    s2[t] = fmaxf((acc - g2m[t]) * s + g2b[t], 0.f);
  }
  __syncthreads();
  if (t < 2) {
    float acc = 0.f;
    const float* w = g3 + (size_t)t * H2_;
    for (int d = 0; d < H2_; ++d) acc += s2[d] * w[d];
    sgl[t] = acc;
    gl[b * 2 + t] = acc;
  }
  __syncthreads();
  if (t == 0) {
    float l = sgl[1] - sgl[0];
    logit[b] = l;
    prob[b]  = 1.f / (1.f + expf(-l));
  }
}

// ---------------------------------------------------------------------------
// launch
// ---------------------------------------------------------------------------
extern "C" void kernel_launch(void* const* d_in, const int* in_sizes, int n_in,
                              void* d_out, int out_size, void* d_ws, size_t ws_size,
                              hipStream_t stream) {
  const float* q_tok = (const float*)d_in[0];
  const float* p_tok = (const float*)d_in[1];
  const int*   lidx  = (const int*)d_in[2];
  const float* sbn_g = (const float*)d_in[3];
  const float* sbn_b = (const float*)d_in[4];
  const float* sbn_m = (const float*)d_in[5];
  const float* sbn_v = (const float*)d_in[6];
  const float* w1    = (const float*)d_in[7];
  const float* bn1_g = (const float*)d_in[8];
  const float* bn1_b = (const float*)d_in[9];
  const float* bn1_m = (const float*)d_in[10];
  const float* bn1_v = (const float*)d_in[11];
  const float* wt1   = (const float*)d_in[12];
  const float* bt1   = (const float*)d_in[13];
  const float* w2    = (const float*)d_in[14];
  const float* bn2_g = (const float*)d_in[15];
  const float* bn2_b = (const float*)d_in[16];
  const float* bn2_m = (const float*)d_in[17];
  const float* bn2_v = (const float*)d_in[18];
  const float* wt2   = (const float*)d_in[19];
  const float* bt2   = (const float*)d_in[20];
  const float* w3    = (const float*)d_in[21];
  const float* b3    = (const float*)d_in[22];
  const float* g1    = (const float*)d_in[23];
  const float* g1g   = (const float*)d_in[24];
  const float* g1b   = (const float*)d_in[25];
  const float* g1m   = (const float*)d_in[26];
  const float* g1v   = (const float*)d_in[27];
  const float* g2    = (const float*)d_in[28];
  const float* g2g   = (const float*)d_in[29];
  const float* g2b   = (const float*)d_in[30];
  const float* g2m   = (const float*)d_in[31];
  const float* g2v   = (const float*)d_in[32];
  const float* g3    = (const float*)d_in[33];

  // output layout (tuple concatenated flat, fp32)
  float* out = (float*)d_out;
  float* o_seg    = out;                         // [32,2,240,240]
  float* o_gl     = o_seg    + (size_t)B_ * 2 * IMG_ * IMG_;
  float* o_logit  = o_gl     + B_ * 2;
  float* o_prob   = o_logit  + B_;
  float* o_patch  = o_prob   + B_;               // [32,3600]
  float* o_resid  = o_patch  + (size_t)B_ * 3600;
  float* o_ctx    = o_resid  + (size_t)B_ * P_;  // [32,225,768]
  float* o_align  = o_ctx    + (size_t)B_ * P_ * DIM_;
  float* o_idx    = o_align  + (size_t)B_ * P_ * DIM_;

  // workspace carve-out
  char* wsb = (char*)d_ws;
  size_t off = 0;
  auto carve = [&](size_t bytes) -> void* {
    void* p = wsb + off;
    off += (bytes + 255) & ~(size_t)255;
    return p;
  };
  bf16_t* qn_bf  = (bf16_t*)carve((size_t)B_ * P_ * DIM_ * 2);
  bf16_t* pn_bf  = (bf16_t*)carve((size_t)B_ * M_ * DIM_ * 2);
  bf16_t* w1_bf  = (bf16_t*)carve((size_t)HID_ * K1_ * 2);
  float*  maxsim = (float*)carve((size_t)B_ * P_ * 4);
  int*    argmax = (int*)carve((size_t)B_ * P_ * 4);
  bf16_t* cm_bf  = (bf16_t*)carve((size_t)B_ * DIM_ * P_ * 2);
  bf16_t* im_bf  = (bf16_t*)carve((size_t)B_ * P_ * K1_ * 2);
  float*  h1     = (float*)carve((size_t)B_ * HID_ * P_ * 4);
  float*  ht1    = (float*)carve((size_t)B_ * HID_ * 900 * 4);
  float*  h2     = (float*)carve((size_t)B_ * H2_ * 900 * 4);
  float*  ht2    = (float*)carve((size_t)B_ * H2_ * 3600 * 4);
  float*  low    = (float*)carve((size_t)B_ * 2 * 3600 * 4);
  float*  pooled = (float*)carve((size_t)B_ * DIM_ * 4);

  // 1. normalize tokens -> bf16
  k_normalize<<<B_ * (P_ + M_), 256, 0, stream>>>(q_tok, p_tok, qn_bf, pn_bf);
  // 2. conv1 weights -> bf16
  {
    int n = HID_ * K1_;
    k_f32_to_bf16<<<(n + 255) / 256, 256, 0, stream>>>(w1, w1_bf, n);
  }
  // 3. fused sim GEMM + argmax (WMMA, LDS-staged B)
  k_sim_argmax<<<dim3((P_ + 15) / 16, B_), 256, 0, stream>>>(qn_bf, pn_bf, maxsim, argmax);
  // 4. context / aligned / residual / indices + seg-head input
  k_context<<<dim3(P_, B_), 256, 0, stream>>>(q_tok, p_tok, lidx,
                                              sbn_g, sbn_b, sbn_m, sbn_v,
                                              maxsim, argmax,
                                              o_ctx, o_align, o_idx, o_resid, cm_bf);
  // 5a. im2col for conv1
  {
    size_t n = (size_t)B_ * P_ * K1_;
    k_im2col<<<(unsigned)((n + 255) / 256), 256, 0, stream>>>(cm_bf, im_bf);
  }
  // 5b. conv1 GEMM WMMA + BN + ReLU
  k_conv1_wmma<<<dim3((P_ + 15) / 16, HID_ / 16, B_), 32, 0, stream>>>(
      w1_bf, im_bf, bn1_g, bn1_b, bn1_m, bn1_v, h1);
  // 6. convT1 15->30
  {
    int n = B_ * HID_ * 900;
    k_convT1<<<(n + 255) / 256, 256, 0, stream>>>(h1, wt1, bt1, ht1);
  }
  // 7. conv2 + BN + ReLU
  {
    int n = B_ * H2_ * 900;
    k_conv2<<<(n + 255) / 256, 256, 0, stream>>>(ht1, w2, bn2_g, bn2_b, bn2_m, bn2_v, h2);
  }
  // 8. convT2 30->60
  {
    int n = B_ * H2_ * 3600;
    k_convT2<<<(n + 255) / 256, 256, 0, stream>>>(h2, wt2, bt2, ht2);
  }
  // 9. conv3 1x1 + patch_map
  {
    int n = B_ * 3600;
    k_conv3<<<(n + 255) / 256, 256, 0, stream>>>(ht2, w3, b3, low, o_patch);
  }
  // 10. bilinear 60->240
  {
    int n = B_ * 2 * IMG_ * IMG_;
    k_resize<<<(n + 255) / 256, 256, 0, stream>>>(low, o_seg);
  }
  // 11. pooling (avg + top-k residual)
  k_pool_topk<<<B_, 256, 0, stream>>>(o_ctx, o_resid, pooled);
  // 12. MLP head
  k_mlp<<<B_, 128, 0, stream>>>(pooled,
                                g1, g1g, g1b, g1m, g1v,
                                g2, g2g, g2b, g2m, g2v,
                                g3, o_gl, o_logit, o_prob);
}
